// LSTMModel_54039278518964
// MI455X (gfx1250) — matrix operand order repaired and checked by reading, not compiled, not run on hardware
//
#include <hip/hip_runtime.h>

typedef __attribute__((ext_vector_type(16))) __bf16 v16bf;
typedef __attribute__((ext_vector_type(8)))  float  v8f;
typedef unsigned int u32;
typedef __attribute__((ext_vector_type(4))) u32 u32x4;
typedef unsigned short u16;

union FragU { v16bf v; u32x4 q[2]; };

__device__ __forceinline__ u16 f2bf(float f) {
  u32 u = __float_as_uint(f);
  u32 r = u + 0x7FFFu + ((u >> 16) & 1u);   // round-to-nearest-even
  return (u16)(r >> 16);
}

__device__ __forceinline__ float sigm(float x) { return 1.0f / (1.0f + __expf(-x)); }

// A fragment: 16(M) x 32(K) bf16, row-major source with leading dim ld.
// ISA layout: lane&15 = M row; lane>>4 selects K-halves {0..7,16..23} vs {8..15,24..31}.
__device__ __forceinline__ v16bf load_a(const u16* __restrict__ base, int ld, int kk, int lane) {
  int r = lane & 15, kh = lane >> 4;
  const u16* p = base + (size_t)r * ld + kk + kh * 8;
  FragU u;
  u.q[0] = *(const u32x4*)p;          // K = kk+8*kh .. +7
  u.q[1] = *(const u32x4*)(p + 16);   // K = kk+16+8*kh .. +7
  return u.v;
}

// B fragment: 32(K) x 16(N) from row-major weights W[n][k] (so GEMM computes A*W^T).
// ISA layout: lane&15 = N col; lane>>4 selects K-halves {0..7,16..23} vs {8..15,24..31}.
__device__ __forceinline__ v16bf load_b(const u16* base, int ld, int kk, int lane) {
  int c = lane & 15, kh = lane >> 4;
  const u16* p = base + (size_t)c * ld + kk + kh * 8;
  FragU u;
  u.q[0] = *(const u32x4*)p;          // K = kk+8*kh .. +7
  u.q[1] = *(const u32x4*)(p + 16);   // K = kk+16+8*kh .. +7
  return u.v;
}

// ---------------------------------------------------------------------------
// Persistent fused LSTM layer. 16 WGs x 256 threads. WG n owns hidden slice
// [n*32, n*32+32) => 128 gate rows of [W_ih | W_hh] staged in LDS once.
// Per step: gates = bias + [x_s ; h_{s-1}] x [W_ih ; W_hh]^T via WMMA,
// cell state lives in VGPRs, h ping-pongs through L2, grid barrier per step.
// ---------------------------------------------------------------------------
__global__ void __launch_bounds__(256)
k_lstm_layer(const u16* __restrict__ xin, int Din,
             const u16* __restrict__ wih, const u16* __restrict__ whh,
             const float* __restrict__ b_ih, const float* __restrict__ b_hh,
             u16* __restrict__ hping, u16* __restrict__ hpong,
             u16* __restrict__ hall, int store_all, int S,
             unsigned* cnt, unsigned* gen)
{
  const int H = 512, Bm = 64;
  const int Ktot = Din + H;
  const int ldw  = Ktot + 8;              // +16B pad per row: conflict-free ds_load_b128
  extern __shared__ u16 wlds[];           // [128][ldw] bf16 (194..258 KB, CDNA5 WGP LDS)

  const int n   = blockIdx.x;             // hidden slice owner
  const int j0  = n * 32;
  const int tid = threadIdx.x;
  const unsigned nwg = gridDim.x;

  // ---- stage this WG's 128 weight rows (4 gates x 32 hidden) into LDS ----
  const int chunks = Ktot >> 3;           // 16B chunks per row
  for (int idx = tid; idx < 128 * chunks; idx += 256) {
    int lr = idx / chunks;
    int ko = (idx - lr * chunks) << 3;
    int g = lr >> 5, r = lr & 31;
    int grow = g * H + j0 + r;
    u32x4 v;
    if (ko < Din) v = *(const u32x4*)(wih + (size_t)grow * Din + ko);
    else          v = *(const u32x4*)(whh + (size_t)grow * H + (ko - Din));
    *(u32x4*)(wlds + (size_t)lr * ldw + ko) = v;
  }
  __syncthreads();

  const int wave  = tid >> 5, lane = tid & 31;
  const int m0    = (wave & 3) * 16;      // batch tile
  const int jloc  = (wave >> 2) * 16;     // 16-wide hidden sub-patch in slice
  const int col   = j0 + jloc + (lane & 15);
  const int khalf = lane >> 4;

  float bias[4];
#pragma unroll
  for (int g = 0; g < 4; ++g) bias[g] = b_ih[g * H + col] + b_hh[g * H + col];

  v8f c;                                   // cell state: resident in VGPRs all 512 steps
#pragma unroll
  for (int e = 0; e < 8; ++e) c[e] = 0.0f;

  for (int s = 0; s < S; ++s) {
    const u16* hcur = (s & 1) ? hpong : hping;
    u16*       hnxt = (s & 1) ? hping : hpong;

    v8f acc[4];
#pragma unroll
    for (int g = 0; g < 4; ++g)
#pragma unroll
      for (int e = 0; e < 8; ++e) acc[g][e] = bias[g];

    // phase 1: input contribution (K over Din)
    const u16* xa = xin + (size_t)s * Bm * Din + (size_t)m0 * Din;
    for (int kk = 0; kk < Din; kk += 32) {
      v16bf a = load_a(xa, Din, kk, lane);
#pragma unroll
      for (int g = 0; g < 4; ++g) {
        v16bf b = load_b(wlds + (size_t)(g * 32 + jloc) * ldw, ldw, kk, lane);
        acc[g] = __builtin_amdgcn_wmma_f32_16x16x32_bf16(false, a, false, b,
                                                         (short)0, acc[g], false, false);
      }
    }
    // phase 2: recurrent contribution (K over H)
    const u16* ha = hcur + (size_t)m0 * H;
    for (int kk = 0; kk < H; kk += 32) {
      v16bf a = load_a(ha, H, kk, lane);
#pragma unroll
      for (int g = 0; g < 4; ++g) {
        v16bf b = load_b(wlds + (size_t)(g * 32 + jloc) * ldw + Din, ldw, kk, lane);
        acc[g] = __builtin_amdgcn_wmma_f32_16x16x32_bf16(false, a, false, b,
                                                         (short)0, acc[g], false, false);
      }
    }

    if (s + 1 < S)  // warm next step's activations (global_prefetch_b8)
      __builtin_prefetch(xin + (size_t)(s + 1) * Bm * Din
                             + (size_t)(m0 + (lane & 15)) * Din, 0, 1);

    // ---- elementwise LSTM cell update (8 batch rows per lane) ----
    v8f h;
#pragma unroll
    for (int e = 0; e < 8; ++e) {
      float iv = sigm(acc[0][e]);
      float fv = sigm(acc[1][e]);
      float gv = tanhf(acc[2][e]);
      float ov = sigm(acc[3][e]);
      float cv = fv * c[e] + iv * gv;
      c[e] = cv;
      h[e] = ov * tanhf(cv);
    }

#pragma unroll
    for (int e = 0; e < 8; ++e) {
      int m = m0 + e + 8 * khalf;
      u16 hb = f2bf(h[e]);
      hnxt[(size_t)m * H + col] = hb;
      if (store_all) hall[(size_t)s * Bm * H + (size_t)m * H + col] = hb;
    }

    // ---- device-wide step barrier (sense-reversal, s_sleep spin) ----
    __syncthreads();
    if (tid == 0) {
      __threadfence();
      unsigned g0 = __hip_atomic_load(gen, __ATOMIC_RELAXED, __HIP_MEMORY_SCOPE_AGENT);
      unsigned prev = __hip_atomic_fetch_add(cnt, 1u, __ATOMIC_ACQ_REL, __HIP_MEMORY_SCOPE_AGENT);
      if (prev == nwg - 1) {
        __hip_atomic_store(cnt, 0u, __ATOMIC_RELAXED, __HIP_MEMORY_SCOPE_AGENT);
        __hip_atomic_fetch_add(gen, 1u, __ATOMIC_RELEASE, __HIP_MEMORY_SCOPE_AGENT);
      } else {
        while (__hip_atomic_load(gen, __ATOMIC_ACQUIRE, __HIP_MEMORY_SCOPE_AGENT) == g0)
          __builtin_amdgcn_s_sleep(2);
      }
    }
    __syncthreads();
    __threadfence();   // invalidate near caches so fresh h is observed
  }
}

// Final FC: [64x512] x [128x512]^T + b -> [64x128]. One block, 8 waves, WMMA.
__global__ void __launch_bounds__(256)
k_fc(const u16* __restrict__ hA, const u16* __restrict__ wB,
     const float* __restrict__ bias, float* __restrict__ out)
{
  const int H = 512, OUT = 128;
  int tid = threadIdx.x, wave = tid >> 5, lane = tid & 31;
  int m0 = (wave & 3) * 16;
  int nb = (wave >> 2) * 64;
  int khalf = lane >> 4;

  v8f acc[4];
#pragma unroll
  for (int t = 0; t < 4; ++t) {
    float bv = bias[nb + t * 16 + (lane & 15)];
#pragma unroll
    for (int e = 0; e < 8; ++e) acc[t][e] = bv;
  }
  const u16* a0 = hA + (size_t)m0 * H;
  for (int kk = 0; kk < H; kk += 32) {
    v16bf a = load_a(a0, H, kk, lane);
#pragma unroll
    for (int t = 0; t < 4; ++t) {
      v16bf b = load_b(wB + (size_t)(nb + t * 16) * H, H, kk, lane);
      acc[t] = __builtin_amdgcn_wmma_f32_16x16x32_bf16(false, a, false, b,
                                                       (short)0, acc[t], false, false);
    }
  }
#pragma unroll
  for (int t = 0; t < 4; ++t) {
    int coln = nb + t * 16 + (lane & 15);
#pragma unroll
    for (int e = 0; e < 8; ++e) {
      int m = m0 + e + 8 * khalf;
      out[(size_t)m * OUT + coln] = acc[t][e];
    }
  }
}

// fp32 -> bf16 flat cast
__global__ void k_cast_bf16(const float* __restrict__ in, u16* __restrict__ outp, int n) {
  for (int i = blockIdx.x * blockDim.x + threadIdx.x; i < n; i += gridDim.x * blockDim.x)
    outp[i] = f2bf(in[i]);
}

// x [B,S,D] f32 -> time-major xt [S,B,D] bf16
__global__ void k_cvt_x(const float* __restrict__ x, u16* __restrict__ xt) {
  const int B = 64, S = 512, D = 256;
  int total = B * S * D;
  for (int i = blockIdx.x * blockDim.x + threadIdx.x; i < total; i += gridDim.x * blockDim.x) {
    int b = i / (S * D);
    int rem = i - b * (S * D);
    int s = rem / D;
    int d = rem - s * D;
    xt[(size_t)s * (B * D) + (size_t)b * D + d] = f2bf(x[i]);
  }
}

__global__ void k_zero(u32* p, int n) {
  for (int i = blockIdx.x * blockDim.x + threadIdx.x; i < n; i += gridDim.x * blockDim.x)
    p[i] = 0u;
}

extern "C" void kernel_launch(void* const* d_in, const int* in_sizes, int n_in,
                              void* d_out, int out_size, void* d_ws, size_t ws_size,
                              hipStream_t stream) {
  const int B = 64, S = 512, D = 256, H = 512, OUT = 128;
  const float* x    = (const float*)d_in[0];
  const float* Wih0 = (const float*)d_in[1];
  const float* Whh0 = (const float*)d_in[2];
  const float* bih0 = (const float*)d_in[3];
  const float* bhh0 = (const float*)d_in[4];
  const float* Wih1 = (const float*)d_in[5];
  const float* Whh1 = (const float*)d_in[6];
  const float* bih1 = (const float*)d_in[7];
  const float* bhh1 = (const float*)d_in[8];
  const float* fcwf = (const float*)d_in[9];
  const float* fcb  = (const float*)d_in[10];
  float* out = (float*)d_out;

  // workspace layout (all offsets 256B-aligned); total ~58 MB
  char* ws = (char*)d_ws;
  unsigned* bar = (unsigned*)ws;                 // [0]=count [1]=generation
  u16* h0p = (u16*)(ws + 256);
  u16* h0q = h0p + B * H;
  u16* h1p = h0q + B * H;
  u16* h1q = h1p + B * H;
  u16* xt  = (u16*)(ws + 262400);                // [S,B,D] bf16
  u16* hall = xt  + (size_t)S * B * D;           // [S,B,H] bf16 (layer0 out / layer1 in)
  u16* wih0 = hall + (size_t)S * B * H;
  u16* whh0 = wih0 + (size_t)4 * H * D;
  u16* wih1 = whh0 + (size_t)4 * H * H;
  u16* whh1 = wih1 + (size_t)4 * H * H;
  u16* fcw  = whh1 + (size_t)4 * H * H;

  // zero barrier + h ping-pong buffers (h0 = 0)
  k_zero<<<64, 256, 0, stream>>>((u32*)ws, (256 + 4 * B * H * 2) / 4);
  k_cvt_x<<<2048, 256, 0, stream>>>(x, xt);
  k_cast_bf16<<<512,  256, 0, stream>>>(Wih0, wih0, 4 * H * D);
  k_cast_bf16<<<1024, 256, 0, stream>>>(Whh0, whh0, 4 * H * H);
  k_cast_bf16<<<1024, 256, 0, stream>>>(Wih1, wih1, 4 * H * H);
  k_cast_bf16<<<1024, 256, 0, stream>>>(Whh1, whh1, 4 * H * H);
  k_cast_bf16<<<64,   256, 0, stream>>>(fcwf, fcw, OUT * H);

  size_t lds0 = (size_t)128 * (D + H + 8) * 2;   // 194 KB
  k_lstm_layer<<<16, 256, lds0, stream>>>(xt, D, wih0, whh0, bih0, bhh0,
                                          h0p, h0q, hall, 1, S, bar, bar + 1);
  size_t lds1 = (size_t)128 * (H + H + 8) * 2;   // 258 KB
  k_lstm_layer<<<16, 256, lds1, stream>>>(hall, H, wih1, whh1, bih1, bhh1,
                                          h1p, h1q, nullptr, 0, S, bar, bar + 1);
  // last write of layer1 (s=511, odd) lands in ping buffer
  k_fc<<<1, 256, 0, stream>>>(h1p, fcw, fcb, out);

  (void)in_sizes; (void)n_in; (void)out_size; (void)ws_size;
}
